// LigerLMHeadKTO_45131516346681
// MI455X (gfx1250) — matrix operand order repaired and checked by reading, not compiled, not run on hardware
//
#include <hip/hip_runtime.h>

typedef __attribute__((ext_vector_type(16))) __bf16   v16bf;
typedef __attribute__((ext_vector_type(2)))  __bf16   v2bf;
typedef __attribute__((ext_vector_type(8)))  float    v8f;
typedef __attribute__((ext_vector_type(4)))  float    f32x4;
typedef __attribute__((ext_vector_type(8)))  unsigned u32x8;
typedef __attribute__((ext_vector_type(4)))  unsigned u32x4;

#define HDIM   2048
#define VOCAB  32000
#define NROWS  2048            // B*T
#define TLEN   512
#define MBLK   64
#define NBLK   256
#define NCHUNK (VOCAB / NBLK)  // 125
#define KSTEP  32
#define IGNORE (-100LL)

// ---------------------------------------------------------------- conversion
// two f32 -> packed bf16 (one VALU op: v_cvt_pk_bf16_f32)
static __device__ __forceinline__ unsigned cvt2(float lo, float hi) {
#if __has_builtin(__builtin_amdgcn_cvt_pk_bf16_f32)
  v2bf r = __builtin_amdgcn_cvt_pk_bf16_f32(lo, hi);
  return __builtin_bit_cast(unsigned, r);
#else
  v2bf r;
  r[0] = (__bf16)lo;
  r[1] = (__bf16)hi;
  return __builtin_bit_cast(unsigned, r);
#endif
}

// B operand (f32 source): 16 consecutive f32 (K ascending) -> v16bf
static __device__ __forceinline__ v16bf cvtB(const float* __restrict__ p) {
  f32x4 f0 = *(const f32x4*)(p);
  f32x4 f1 = *(const f32x4*)(p + 4);
  f32x4 f2 = *(const f32x4*)(p + 8);
  f32x4 f3 = *(const f32x4*)(p + 12);
  u32x8 u;
  u[0] = cvt2(f0[0], f0[1]); u[1] = cvt2(f0[2], f0[3]);
  u[2] = cvt2(f1[0], f1[1]); u[3] = cvt2(f1[2], f1[3]);
  u[4] = cvt2(f2[0], f2[1]); u[5] = cvt2(f2[2], f2[3]);
  u[6] = cvt2(f3[0], f3[1]); u[7] = cvt2(f3[2], f3[3]);
  return __builtin_bit_cast(v16bf, u);
}

// A operand (f32 source): lane-half holds K = koff+0..15
static __device__ __forceinline__ v16bf cvtA(const float* __restrict__ p) {
  f32x4 f0 = *(const f32x4*)(p);
  f32x4 f1 = *(const f32x4*)(p + 4);
  f32x4 f2 = *(const f32x4*)(p + 8);
  f32x4 f3 = *(const f32x4*)(p + 12);
  u32x8 u;
  u[0] = cvt2(f0[0], f0[1]); u[1] = cvt2(f0[2], f0[3]);
  u[2] = cvt2(f1[0], f1[1]); u[3] = cvt2(f1[2], f1[3]);
  u[4] = cvt2(f2[0], f2[1]); u[5] = cvt2(f2[2], f2[3]);
  u[6] = cvt2(f3[0], f3[1]); u[7] = cvt2(f3[2], f3[3]);
  return __builtin_bit_cast(v16bf, u);
}

// bf16 operand loads (pre-converted path, no VALU work)
static __device__ __forceinline__ v16bf loadA16(const __bf16* __restrict__ p) {
  u32x4 lo = *(const u32x4*)(p);        // K koff+0..7
  u32x4 hi = *(const u32x4*)(p + 8);    // K koff+8..15
  u32x8 u;
  u[0] = lo[0]; u[1] = lo[1]; u[2] = lo[2]; u[3] = lo[3];
  u[4] = hi[0]; u[5] = hi[1]; u[6] = hi[2]; u[7] = hi[3];
  return __builtin_bit_cast(v16bf, u);
}
static __device__ __forceinline__ v16bf loadB16(const __bf16* __restrict__ p) {
  u32x8 u = *(const u32x8*)(p);         // 16 consecutive bf16 (32B, aligned)
  return __builtin_bit_cast(v16bf, u);
}

// bulk f32 -> bf16 conversion, 8 elements/thread, b128 in / b128 out
__global__ __launch_bounds__(256) void cvt_f32_to_bf16(
    const float* __restrict__ src, __bf16* __restrict__ dst, int n8)
{
  int i = blockIdx.x * blockDim.x + threadIdx.x;
  if (i >= n8) return;
  const f32x4* s = (const f32x4*)(src + (size_t)i * 8);
  f32x4 f0 = s[0], f1 = s[1];
  u32x4 u;
  u[0] = cvt2(f0[0], f0[1]); u[1] = cvt2(f0[2], f0[3]);
  u[2] = cvt2(f1[0], f1[1]); u[3] = cvt2(f1[2], f1[3]);
  *(u32x4*)(dst + (size_t)i * 8) = u;
}

// ------------------------------------------------------- shared lse epilogue
static __device__ __forceinline__ void lse_epilogue(
    const v8f (&acc)[2][4],
    int rowTile, int colWave, int mrow0, int wm, int wn,
    int tid, int lane, int model, int chunk,
    const long long* __restrict__ y,
    float* __restrict__ pmax, float* __restrict__ psum,
    float* __restrict__ tok)
{
  const int half = lane >> 4;
  __shared__ float pM[4][16][4];
  __shared__ float pS[4][16][4];

#pragma unroll
  for (int t = 0; t < 2; ++t) {
    float rmax[8], rsum[8];
#pragma unroll
    for (int r = 0; r < 8; ++r) {
      float v0 = acc[t][0][r], v1 = acc[t][1][r];
      float v2 = acc[t][2][r], v3 = acc[t][3][r];
      float lm = fmaxf(fmaxf(v0, v1), fmaxf(v2, v3));
#pragma unroll
      for (int m = 1; m < 16; m <<= 1) lm = fmaxf(lm, __shfl_xor(lm, m, 32));
      float ls = __expf(v0 - lm) + __expf(v1 - lm) +
                 __expf(v2 - lm) + __expf(v3 - lm);
#pragma unroll
      for (int m = 1; m < 16; m <<= 1) ls += __shfl_xor(ls, m, 32);
      rmax[r] = lm; rsum[r] = ls;
    }

    // target-logit gather: exactly one lane in the whole grid matches per row
#pragma unroll
    for (int r = 0; r < 8; ++r) {
      int grow = mrow0 + t * 16 + half * 8 + r;
      long long tg = y[grow];
#pragma unroll
      for (int c = 0; c < 4; ++c) {
        int gcol = colWave + c * 16 + (lane & 15);
        if ((long long)gcol == tg) tok[(size_t)model * NROWS + grow] = acc[t][c][r];
      }
    }

    if ((lane & 15) == 0) {
      int mt = wm * 2 + t;
#pragma unroll
      for (int r = 0; r < 8; ++r) {
        pM[mt][half * 8 + r][wn] = rmax[r];
        pS[mt][half * 8 + r][wn] = rsum[r];
      }
    }
  }

  __syncthreads();
  if (tid < 64) {
    int mt = tid >> 4, m = tid & 15;
    float M = -3.4e38f;
#pragma unroll
    for (int i = 0; i < 4; ++i) M = fmaxf(M, pM[mt][m][i]);
    float S = 0.f;
#pragma unroll
    for (int i = 0; i < 4; ++i) S += pS[mt][m][i] * __expf(pM[mt][m][i] - M);
    int grow = rowTile + mt * 16 + m;
    size_t o = ((size_t)model * NROWS + grow) * NCHUNK + chunk;
    pmax[o] = M;
    psum[o] = S;
  }
}

// ------------------------------------------- GEMM variant 1: fused f32->bf16
__global__ __launch_bounds__(256) void gemm_lse_f32(
    const float* __restrict__ x, const float* __restrict__ ref_x,
    const float* __restrict__ W, const float* __restrict__ ref_W,
    const long long* __restrict__ y,
    float* __restrict__ pmax, float* __restrict__ psum,
    float* __restrict__ tok)
{
  const int rowTile = blockIdx.x * MBLK;      // fastest dim: L2-reuse of W chunk
  const int col0    = blockIdx.y * NBLK;
  const int model   = blockIdx.z;
  const float* A  = model ? ref_x : x;
  const float* Bm = model ? ref_W : W;

  const int tid  = threadIdx.x;
  const int w    = tid >> 5;
  const int lane = tid & 31;
  const int wm   = w >> 2;          // 0..1  (M) -> 32 rows/wave
  const int wn   = w & 3;           // 0..3  (N) -> 64 cols/wave

  const int mrow0   = rowTile + wm * 32;
  const int colWave = col0 + wn * 64;

  const int am = lane & 15;
  const int ak = (lane >> 4) * 16;
  const float* aptr0 = A + (size_t)(mrow0 + am) * HDIM + ak;
  const float* aptr1 = A + (size_t)(mrow0 + 16 + am) * HDIM + ak;

  const int bn = lane & 15;
  const int bk = (lane >> 4) * 16;
  const float* bbase = Bm + (size_t)(colWave + bn) * HDIM + bk;

  v8f acc[2][4];
#pragma unroll
  for (int t = 0; t < 2; ++t)
#pragma unroll
    for (int c = 0; c < 4; ++c) acc[t][c] = v8f{};

  for (int k = 0; k < HDIM; k += KSTEP) {
    __builtin_prefetch(aptr0 + k + KSTEP, 0, 3);
    __builtin_prefetch(aptr1 + k + KSTEP, 0, 3);
    v16bf a0 = cvtA(aptr0 + k);
    v16bf a1 = cvtA(aptr1 + k);
#pragma unroll
    for (int c = 0; c < 4; ++c) {
      const float* bp = bbase + (size_t)(c * 16) * HDIM + k;
      __builtin_prefetch(bp + KSTEP, 0, 3);
      v16bf b = cvtB(bp);
      acc[0][c] = __builtin_amdgcn_wmma_f32_16x16x32_bf16(
          false, a0, false, b, (short)0, acc[0][c], false, false);
      acc[1][c] = __builtin_amdgcn_wmma_f32_16x16x32_bf16(
          false, a1, false, b, (short)0, acc[1][c], false, false);
    }
  }

  lse_epilogue(acc, rowTile, colWave, mrow0, wm, wn, tid, lane, model,
               blockIdx.y, y, pmax, psum, tok);
}

// --------------------------------------- GEMM variant 2: pre-converted bf16
__global__ __launch_bounds__(256) void gemm_lse_bf16(
    const __bf16* __restrict__ xb,      // [2][NROWS][HDIM]
    const __bf16* __restrict__ Wb,      // [2][VOCAB][HDIM]
    const long long* __restrict__ y,
    float* __restrict__ pmax, float* __restrict__ psum,
    float* __restrict__ tok)
{
  const int rowTile = blockIdx.x * MBLK;
  const int col0    = blockIdx.y * NBLK;
  const int model   = blockIdx.z;
  const __bf16* A  = xb + (size_t)model * NROWS * HDIM;
  const __bf16* Bm = Wb + (size_t)model * VOCAB * HDIM;

  const int tid  = threadIdx.x;
  const int w    = tid >> 5;
  const int lane = tid & 31;
  const int wm   = w >> 2;
  const int wn   = w & 3;

  const int mrow0   = rowTile + wm * 32;
  const int colWave = col0 + wn * 64;

  const int am = lane & 15;
  const int ak = (lane >> 4) * 16;
  const __bf16* aptr0 = A + (size_t)(mrow0 + am) * HDIM + ak;
  const __bf16* aptr1 = A + (size_t)(mrow0 + 16 + am) * HDIM + ak;

  const int bn = lane & 15;
  const int bk = (lane >> 4) * 16;
  const __bf16* bbase = Bm + (size_t)(colWave + bn) * HDIM + bk;

  v8f acc[2][4];
#pragma unroll
  for (int t = 0; t < 2; ++t)
#pragma unroll
    for (int c = 0; c < 4; ++c) acc[t][c] = v8f{};

  for (int k = 0; k < HDIM; k += KSTEP) {
    __builtin_prefetch(aptr0 + k + KSTEP, 0, 3);
    __builtin_prefetch(aptr1 + k + KSTEP, 0, 3);
    v16bf a0 = loadA16(aptr0 + k);
    v16bf a1 = loadA16(aptr1 + k);
#pragma unroll
    for (int c = 0; c < 4; ++c) {
      const __bf16* bp = bbase + (size_t)(c * 16) * HDIM + k;
      __builtin_prefetch(bp + KSTEP, 0, 3);
      v16bf b = loadB16(bp);
      acc[0][c] = __builtin_amdgcn_wmma_f32_16x16x32_bf16(
          false, a0, false, b, (short)0, acc[0][c], false, false);
      acc[1][c] = __builtin_amdgcn_wmma_f32_16x16x32_bf16(
          false, a1, false, b, (short)0, acc[1][c], false, false);
    }
  }

  lse_epilogue(acc, rowTile, colWave, mrow0, wm, wn, tid, lane, model,
               blockIdx.y, y, pmax, psum, tok);
}

// ----------------------------------------------------------- reduce kernels
__global__ __launch_bounds__(256) void reduce_lse_kernel(
    const float* __restrict__ pmax, const float* __restrict__ psum,
    const float* __restrict__ tok, float* __restrict__ ptl)
{
  int idx = blockIdx.x * blockDim.x + threadIdx.x;
  if (idx >= 2 * NROWS) return;
  const float* pm = pmax + (size_t)idx * NCHUNK;
  const float* ps = psum + (size_t)idx * NCHUNK;
  float M = -3.4e38f;
  for (int c = 0; c < NCHUNK; ++c) M = fmaxf(M, pm[c]);
  float S = 0.f;
  for (int c = 0; c < NCHUNK; ++c) S += ps[c] * __expf(pm[c] - M);
  ptl[idx] = tok[idx] - (M + __logf(S));
}

__global__ __launch_bounds__(256) void final_loss_kernel(
    const float* __restrict__ ptl, const long long* __restrict__ y,
    float* __restrict__ out)
{
  __shared__ float avg[8];
  int t = threadIdx.x;
  int p = t >> 5, lane = t & 31;
  int model = p >> 2, b = p & 3;
  const float* base = ptl + (size_t)model * NROWS + b * TLEN;
  const long long* yb = y + b * TLEN;
  float s = 0.f, cnt = 0.f;
  for (int i = lane; i < TLEN; i += 32) {
    if (yb[i] != IGNORE) { s += base[i]; cnt += 1.f; }
  }
#pragma unroll
  for (int m = 16; m >= 1; m >>= 1) {
    s   += __shfl_xor(s, m, 32);
    cnt += __shfl_xor(cnt, m, 32);
  }
  if (lane == 0) avg[p] = (cnt > 0.f) ? (s / cnt) : 0.f;
  __syncthreads();
  if (t == 0) {
    const float beta = 0.1f;
    float loss = 0.f;
    for (int i = 0; i < 2; ++i) {             // chosen
      float lr = avg[i] - avg[4 + i];
      loss += 1.f - 1.f / (1.f + __expf(-beta * lr));
    }
    for (int i = 2; i < 4; ++i) {             // rejected
      float lr = avg[i] - avg[4 + i];
      loss += 1.f - 1.f / (1.f + __expf(beta * lr));
    }
    out[0] = loss * 0.5f;
  }
}

// ------------------------------------------------------------------- launch
extern "C" void kernel_launch(void* const* d_in, const int* in_sizes, int n_in,
                              void* d_out, int out_size, void* d_ws, size_t ws_size,
                              hipStream_t stream) {
  const float*     x     = (const float*)d_in[0];
  const float*     ref_x = (const float*)d_in[1];
  const long long* y     = (const long long*)d_in[2];
  const float*     W     = (const float*)d_in[3];
  const float*     ref_W = (const float*)d_in[4];

  // partials region (always present)
  float* wsf  = (float*)d_ws;
  float* pmax = wsf;
  float* psum = pmax + (size_t)2 * NROWS * NCHUNK;
  float* tok  = psum + (size_t)2 * NROWS * NCHUNK;
  float* ptl  = tok  + (size_t)2 * NROWS;
  const size_t partial_bytes =
      ((size_t)2 * NROWS * NCHUNK * 2 + (size_t)2 * NROWS * 2) * sizeof(float);

  // optional bf16 staging region
  const size_t x_elems = (size_t)NROWS * HDIM;     // per model
  const size_t w_elems = (size_t)VOCAB * HDIM;     // per model
  const size_t bf16_bytes = 2 * (x_elems + w_elems) * sizeof(__bf16);

  dim3 grid(NROWS / MBLK, NCHUNK, 2);   // 32 x 125 x 2

  if (ws_size >= partial_bytes + bf16_bytes) {
    // Path A: pre-convert to bf16 once, then conversion-free WMMA GEMM
    __bf16* xb = (__bf16*)((char*)d_ws + partial_bytes);
    __bf16* Wb = xb + 2 * x_elems;
    const int xn8 = (int)(x_elems / 8), wn8 = (int)(w_elems / 8);
    cvt_f32_to_bf16<<<(xn8 + 255) / 256, 256, 0, stream>>>(x,     xb,            xn8);
    cvt_f32_to_bf16<<<(xn8 + 255) / 256, 256, 0, stream>>>(ref_x, xb + x_elems,  xn8);
    cvt_f32_to_bf16<<<(wn8 + 255) / 256, 256, 0, stream>>>(W,     Wb,            wn8);
    cvt_f32_to_bf16<<<(wn8 + 255) / 256, 256, 0, stream>>>(ref_W, Wb + w_elems,  wn8);
    gemm_lse_bf16<<<grid, dim3(256), 0, stream>>>(xb, Wb, y, pmax, psum, tok);
  } else {
    // Path B: fused conversion inside the GEMM
    gemm_lse_f32<<<grid, dim3(256), 0, stream>>>(x, ref_x, W, ref_W, y,
                                                 pmax, psum, tok);
  }

  reduce_lse_kernel<<<(2 * NROWS + 255) / 256, 256, 0, stream>>>(pmax, psum,
                                                                 tok, ptl);
  final_loss_kernel<<<1, 256, 0, stream>>>(ptl, y, (float*)d_out);
}
